// MainModule_3461743640710
// MI455X (gfx1250) — matrix operand order repaired and checked by reading, not compiled, not run on hardware
//
#include <hip/hip_runtime.h>
#include <math.h>

// ---------------------------------------------------------------------------
// MI455X (gfx1250, wave32) implementation of the trend+stack LSTM reference.
//  - f16 WMMA (v_wmma_f32_16x16x32_f16) for input-projection GEMMs and the
//    recurrent GEMV (h broadcast across B columns).
//  - Weights converted once per launch to f16 in workspace (80 MiB, L2-resident).
//  - Recurrent scan: 16 persistent workgroups, global generation barrier.
// ---------------------------------------------------------------------------

typedef __attribute__((ext_vector_type(16))) _Float16 v16h;
typedef __attribute__((ext_vector_type(8)))  _Float16 v8h;
typedef __attribute__((ext_vector_type(8)))  float    v8f;

#define HDIM 1024     // hidden size (H1 == H2)
#define GDIM 4096     // 4*H gate rows
#define T1   4096     // trend sequence length
#define T2   4097     // stacked sequence length
#define TP   4112     // padded sequence length (257 * 16)
#define CCH  64       // input channels

__device__ __forceinline__ float sigmoidf_(float x) { return 1.0f / (1.0f + __expf(-x)); }

// ---------------------------------------------------------------------------
// utility kernels
// ---------------------------------------------------------------------------
__global__ void zero_u32(unsigned* __restrict__ p, size_t n) {
    for (size_t i = blockIdx.x * (size_t)blockDim.x + threadIdx.x; i < n;
         i += (size_t)gridDim.x * blockDim.x)
        p[i] = 0u;
}

__global__ void cvt_f16(const float* __restrict__ s, _Float16* __restrict__ d, size_t n) {
    for (size_t i = blockIdx.x * (size_t)blockDim.x + threadIdx.x; i < n;
         i += (size_t)gridDim.x * blockDim.x)
        d[i] = (_Float16)s[i];
}

// trend LSTM input: avgpool over a singleton axis collapses to x/3
__global__ void trend_input(const float* __restrict__ x0, float* __restrict__ out) {
    int t = blockIdx.x * blockDim.x + threadIdx.x;
    if (t >= T1) return;
    out[t] = x0[(size_t)t * CCH] * (1.0f / 3.0f);
}

// Gin[t][j] = x[t] * Wih[j,0] + bih[j] + bhh[j]   (input dim == 1 layers)
__global__ void gin_scalar(const float* __restrict__ xin, const float* __restrict__ Wih,
                           const float* __restrict__ bih, const float* __restrict__ bhh,
                           float* __restrict__ Gin, int T) {
    size_t n = (size_t)T * GDIM;
    for (size_t i = blockIdx.x * (size_t)blockDim.x + threadIdx.x; i < n;
         i += (size_t)gridDim.x * blockDim.x) {
        int t = (int)(i / GDIM);
        int j = (int)(i % GDIM);
        Gin[i] = xin[t] * Wih[j] + bih[j] + bhh[j];
    }
}

// ---------------------------------------------------------------------------
// WMMA input-projection GEMM: Gin = X(f16, TPxH) @ W(f16, GxH)^T + (bih+bhh)
// One 16x16 C-tile per wave.  A layout: lane%16 = M row, lane/16 selects the
// K group of 16 (contiguous in the X row); B layout: lane%16 = N column,
// lane/16 selects K group of 16 (contiguous in the W row).
// ---------------------------------------------------------------------------
__global__ __launch_bounds__(256) void wmma_gin_gemm(
    const _Float16* __restrict__ X, const _Float16* __restrict__ W,
    const float* __restrict__ bih, const float* __restrict__ bhh,
    float* __restrict__ Gin, int mTiles) {
    int lane  = threadIdx.x & 31;
    int wave  = threadIdx.x >> 5;
    int tile  = blockIdx.x * 8 + wave;
    int total = mTiles * (GDIM / 16);
    if (tile >= total) return;
    int mTile = tile / (GDIM / 16);
    int nTile = tile % (GDIM / 16);
    int laneM = lane & 15;
    int kg    = lane >> 4;

    const _Float16* Arow = X + (size_t)(mTile * 16 + laneM) * HDIM;
    const _Float16* Brow = W + (size_t)(nTile * 16 + laneM) * HDIM; // column n == W row n

    v8f acc = {};
    for (int k0 = 0; k0 < HDIM; k0 += 32) {
        v16h A = *(const v16h*)(Arow + k0 + kg * 16);
        v16h B = *(const v16h*)(Brow + k0 + kg * 16);
        acc = __builtin_amdgcn_wmma_f32_16x16x32_f16(false, A, false, B,
                                                     (short)0, acc, false, false);
    }
    int n      = nTile * 16 + laneM;
    float bias = bih[n] + bhh[n];
    int mBase  = mTile * 16 + kg * 8;
#pragma unroll
    for (int v = 0; v < 8; ++v)
        Gin[(size_t)(mBase + v) * GDIM + n] = acc[v] + bias;
}

// ---------------------------------------------------------------------------
// Recurrent scan: 16 persistent workgroups x 512 threads (16 waves).
// Block b owns hidden slice [b*64, b*64+64): 4 gates x 4 row-tiles = 16 tiles,
// one per wave.  B operand = h chunk broadcast to all 16 columns (all D
// columns identical -> lanes with lane%16==0 commit results).
// h exchanged via 2-entry global ping-pong + generation-counter grid barrier.
// ---------------------------------------------------------------------------
__global__ __launch_bounds__(512) void lstm_scan(
    const float* __restrict__ Gin, const _Float16* __restrict__ Whh,
    _Float16* __restrict__ hsOut, _Float16* __restrict__ hbuf,
    unsigned* __restrict__ barCount, unsigned* __restrict__ barGen, int T) {
    __shared__ alignas(32) _Float16 sh_h[HDIM];
    __shared__ float sh_g[256];
    __shared__ float sh_c[64];

    int tid   = threadIdx.x;
    int lane  = tid & 31;
    int wave  = tid >> 5;
    int b     = blockIdx.x;      // 0..15
    int q     = wave >> 2;       // gate index 0..3 (i,f,g,o)
    int rt    = wave & 3;        // 16-row tile within this block's gate slice
    int laneM = lane & 15;
    int kg    = lane >> 4;
    int rowG  = q * HDIM + b * 64 + rt * 16; // global gate-row base of this wave

    const _Float16* Arow = Whh + (size_t)(rowG + laneM) * HDIM;

    if (tid < 64) sh_c[tid] = 0.0f;

    for (int t = 0; t < T; ++t) {
        const _Float16* hcur = hbuf + (t & 1) * HDIM;
        _Float16*       hnxt = hbuf + ((t + 1) & 1) * HDIM;

        // stage h_{t-1} into LDS (1024 halves = 512 dwords)
        __syncthreads();
        if (tid < HDIM / 2)
            ((unsigned*)sh_h)[tid] = ((const unsigned*)hcur)[tid];
        __syncthreads();

        const float* ginRow = Gin + (size_t)t * GDIM;
        v8f acc = {};
#pragma unroll 4
        for (int k0 = 0; k0 < HDIM; k0 += 32) {
            v16h A = *(const v16h*)(Arow + k0 + kg * 16);
            v16h B = *(const v16h*)(sh_h + k0 + kg * 16); // broadcast h chunk
            acc = __builtin_amdgcn_wmma_f32_16x16x32_f16(false, A, false, B,
                                                         (short)0, acc, false, false);
        }
        if (lane == 0 && t + 1 < T) // pull next step's Gin slice into cache
            __builtin_prefetch(Gin + (size_t)(t + 1) * GDIM + rowG, 0, 1);

        if (laneM == 0) { // column 0 of D: lanes 0 (M 0..7) and 16 (M 8..15)
            int lbase = q * 64 + rt * 16 + kg * 8;
            int gbase = rowG + kg * 8;
#pragma unroll
            for (int v = 0; v < 8; ++v)
                sh_g[lbase + v] = acc[v] + ginRow[gbase + v];
        }
        __syncthreads();

        if (tid < 64) { // elementwise gate math for this block's hidden slice
            float gi = sh_g[tid], gf = sh_g[64 + tid];
            float gg = sh_g[128 + tid], go = sh_g[192 + tid];
            float c  = sigmoidf_(gf) * sh_c[tid] + sigmoidf_(gi) * tanhf(gg);
            float h  = sigmoidf_(go) * tanhf(c);
            sh_c[tid] = c;
            _Float16 h16 = (_Float16)h;
            int J        = b * 64 + tid;
            hnxt[J]      = h16;
            hsOut[(size_t)t * HDIM + J] = h16;
        }

        // grid-wide generation barrier
        __syncthreads();
        if (tid == 0) {
            __threadfence();
            volatile unsigned* vg = (volatile unsigned*)barGen;
            unsigned g = *vg;
            if (atomicAdd(barCount, 1u) + 1u == (unsigned)gridDim.x) {
                *(volatile unsigned*)barCount = 0u;
                __threadfence();
                *vg = g + 1u;
            } else {
                while (*vg == g) __builtin_amdgcn_s_sleep(1);
            }
        }
        __syncthreads();
    }
}

// ---------------------------------------------------------------------------
// regression head: out[row] = dot(hs[row], w) + b   (one wave per row)
// ---------------------------------------------------------------------------
__global__ __launch_bounds__(256) void reg_head(
    const _Float16* __restrict__ hs, const float* __restrict__ w,
    const float* __restrict__ bptr, float* __restrict__ out, int rows) {
    int lane = threadIdx.x & 31;
    int wave = threadIdx.x >> 5;
    int row  = blockIdx.x * 8 + wave;
    if (row >= rows) return;
    const _Float16* hr = hs + (size_t)row * HDIM;
    float s = 0.0f;
    for (int j = lane; j < HDIM; j += 32) s += (float)hr[j] * w[j];
#pragma unroll
    for (int off = 16; off; off >>= 1) s += __shfl_down(s, off, 32);
    if (lane == 0) out[row] = s + bptr[0];
}

// ---------------------------------------------------------------------------
// TSSA: per-column |pairwise-diff| sum and column sums (one block per column)
// ---------------------------------------------------------------------------
__global__ void tssa_reduce(const float* __restrict__ x0, float* __restrict__ tss,
                            float* __restrict__ colsum) {
    int j = blockIdx.x + 1; // columns 1..63 of x0
    __shared__ float r1[256], r2[256];
    float accT = 0.0f, accC = 0.0f;
    for (int t = threadIdx.x; t < T1; t += 256) {
        float a  = x0[(size_t)t * CCH + 0];
        float Bv = x0[(size_t)t * CCH + j];
        float sa = 0.0f, sB = 0.0f;
        if (t > 0) {
            sa = a  - x0[(size_t)(t - 1) * CCH + 0];
            sB = Bv - x0[(size_t)(t - 1) * CCH + j];
        }
        float d  = sa - sB;
        float hp = (sa * sB >= 0.0f) ? d : fabsf(d);
        accT += hp;
        accC += Bv;
    }
    r1[threadIdx.x] = accT;
    r2[threadIdx.x] = accC;
    __syncthreads();
    for (int s = 128; s; s >>= 1) {
        if (threadIdx.x < s) {
            r1[threadIdx.x] += r1[threadIdx.x + s];
            r2[threadIdx.x] += r2[threadIdx.x + s];
        }
        __syncthreads();
    }
    if (threadIdx.x == 0) { tss[blockIdx.x] = r1[0]; colsum[blockIdx.x] = r2[0]; }
}

__global__ void tssa_final(const float* __restrict__ tss, const float* __restrict__ colsum,
                           const float* __restrict__ variance, float* __restrict__ yout) {
    int lane  = threadIdx.x; // 32 threads
    float sig = variance[0];
    float inv = 1.0f / (2.50662827463f * sig); // 1/(sqrt(2*pi)*sig)
    float sT = 0.0f, sTB = 0.0f;
    for (int j = lane; j < 63; j += 32) {
        float tv    = tss[j];
        float score = inv * __expf(-tv * tv / (2.0f * sig * sig));
        float tj    = tanf(score);
        sT  += tj;
        sTB += tj * colsum[j];
    }
#pragma unroll
    for (int off = 16; off; off >>= 1) {
        sT  += __shfl_down(sT, off, 32);
        sTB += __shfl_down(sTB, off, 32);
    }
    if (lane == 0) yout[0] = sTB / sT; // sum(w*Bm) with w = tan(score)/sum(tan)
}

// s_in = [y_tssa, trend...], zero-padded to TP
__global__ void build_seq(const float* __restrict__ y, const float* __restrict__ trend,
                          float* __restrict__ seq) {
    int i = blockIdx.x * blockDim.x + threadIdx.x;
    if (i >= TP) return;
    seq[i] = (i == 0) ? y[0] : (i <= T1 ? trend[i - 1] : 0.0f);
}

// ---------------------------------------------------------------------------
// host-side orchestration
// ---------------------------------------------------------------------------
extern "C" void kernel_launch(void* const* d_in, const int* in_sizes, int n_in,
                              void* d_out, int out_size, void* d_ws, size_t ws_size,
                              hipStream_t stream) {
    (void)in_sizes; (void)n_in; (void)out_size; (void)ws_size;
    const float* x     = (const float*)d_in[0];
    const float* tWih  = (const float*)d_in[1];
    const float* tWhh  = (const float*)d_in[2];
    const float* tBih  = (const float*)d_in[3];
    const float* tBhh  = (const float*)d_in[4];
    const float* tRegW = (const float*)d_in[5];
    const float* tRegB = (const float*)d_in[6];
    const float* var   = (const float*)d_in[7];
    const float* sWih0 = (const float*)d_in[8];
    const float* sWihR = (const float*)d_in[9];
    const float* sWhh  = (const float*)d_in[10];
    const float* sBih  = (const float*)d_in[11];
    const float* sBhh  = (const float*)d_in[12];
    const float* sRegW = (const float*)d_in[13];
    const float* sRegB = (const float*)d_in[14];
    float* out = (float*)d_out;

    char* ws   = (char*)d_ws;
    size_t off = 0;
    auto take = [&](size_t bytes) -> char* {
        char* p = ws + off;
        off += (bytes + 255) & ~(size_t)255;
        return p;
    };
    const size_t WSZ = (size_t)GDIM * HDIM; // 4M elements per weight matrix
    _Float16* whh16  = (_Float16*)take(6 * WSZ * 2); // [trend, s0..s4]
    _Float16* wih16  = (_Float16*)take(4 * WSZ * 2); // stack layers 1..4
    float*    gin    = (float*)take((size_t)TP * GDIM * 4);
    _Float16* hsA    = (_Float16*)take((size_t)TP * HDIM * 2);
    _Float16* hsB    = (_Float16*)take((size_t)TP * HDIM * 2);
    _Float16* hbuf   = (_Float16*)take(2 * HDIM * 2);
    float* trendIn   = (float*)take(T1 * 4);
    float* trend     = (float*)take(T1 * 4);
    float* seqIn     = (float*)take(TP * 4);
    float* tssBuf    = (float*)take(64 * 4);
    float* colsum    = (float*)take(64 * 4);
    float* yscal     = (float*)take(4);
    unsigned* barC   = (unsigned*)take(8);
    unsigned* barG   = (unsigned*)take(8);

    // per-launch init: zero hs buffers (pad rows must be 0), h ping-pong, barrier
    zero_u32<<<2048, 256, 0, stream>>>((unsigned*)hsA, (size_t)TP * HDIM / 2);
    zero_u32<<<2048, 256, 0, stream>>>((unsigned*)hsB, (size_t)TP * HDIM / 2);
    zero_u32<<<4, 256, 0, stream>>>((unsigned*)hbuf, 2 * HDIM / 2);
    zero_u32<<<1, 32, 0, stream>>>(barC, 2); // barrier count only; gen sits 256 B later

    // weight conversion fp32 -> f16 (stays resident in 192MB L2)
    cvt_f16<<<2048, 256, 0, stream>>>(tWhh, whh16, WSZ);
    for (int l = 0; l < 5; ++l)
        cvt_f16<<<2048, 256, 0, stream>>>(sWhh + (size_t)l * WSZ,
                                          whh16 + (size_t)(1 + l) * WSZ, WSZ);
    for (int l = 0; l < 4; ++l)
        cvt_f16<<<2048, 256, 0, stream>>>(sWihR + (size_t)l * WSZ,
                                          wih16 + (size_t)l * WSZ, WSZ);

    // ---- trend branch ----
    trend_input<<<(T1 + 255) / 256, 256, 0, stream>>>(x, trendIn);
    gin_scalar<<<4096, 256, 0, stream>>>(trendIn, tWih, tBih, tBhh, gin, T1);
    lstm_scan<<<16, 512, 0, stream>>>(gin, whh16, hsA, hbuf, barC, barG, T1);
    reg_head<<<(T1 + 7) / 8, 256, 0, stream>>>(hsA, tRegW, tRegB, trend, T1);

    // ---- TSSA branch ----
    tssa_reduce<<<63, 256, 0, stream>>>(x, tssBuf, colsum);
    tssa_final<<<1, 32, 0, stream>>>(tssBuf, colsum, var, yscal);
    build_seq<<<(TP + 255) / 256, 256, 0, stream>>>(yscal, trend, seqIn);

    // ---- stacked LSTM, layer 0 (input dim 1) ----
    zero_u32<<<4, 256, 0, stream>>>((unsigned*)hbuf, 2 * HDIM / 2);
    gin_scalar<<<4096, 256, 0, stream>>>(seqIn, sWih0, sBih, sBhh, gin, T2);
    lstm_scan<<<16, 512, 0, stream>>>(gin, whh16 + WSZ, hsB, hbuf, barC, barG, T2);

    // ---- stacked LSTM, layers 1..4: WMMA GEMM input projection + scan ----
    _Float16* hsPrev = hsB;
    _Float16* hsCur  = hsA;
    for (int l = 1; l < 5; ++l) {
        wmma_gin_gemm<<<(TP / 16) * (GDIM / 16) / 8, 256, 0, stream>>>(
            hsPrev, wih16 + (size_t)(l - 1) * WSZ,
            sBih + (size_t)l * GDIM, sBhh + (size_t)l * GDIM, gin, TP / 16);
        zero_u32<<<4, 256, 0, stream>>>((unsigned*)hbuf, 2 * HDIM / 2);
        lstm_scan<<<16, 512, 0, stream>>>(gin, whh16 + (size_t)(1 + l) * WSZ,
                                          hsCur, hbuf, barC, barG, T2);
        _Float16* tmp = hsPrev; hsPrev = hsCur; hsCur = tmp;
    }

    // ---- output head: (4097,1,1) ----
    reg_head<<<(T2 + 7) / 8, 256, 0, stream>>>(hsPrev, sRegW, sRegB, out, T2);
}
